// MHA_32839319945836
// MI455X (gfx1250) — matrix operand order repaired and checked by reading, compiled, not run on hardware
//
#include <hip/hip_runtime.h>

typedef __attribute__((ext_vector_type(8)))  _Float16 v8h;
typedef __attribute__((ext_vector_type(16))) _Float16 v16h;
typedef __attribute__((ext_vector_type(8)))  float    v8f;

#define HEADS 16
#define DMODEL 1024
#define HEADDIM 64

union H16 { v16h v; v8h h[2]; };

// A-fragment: lane l holds row (l&15); halves of K interleave in 8-blocks:
// elements 0..7 = K kb..kb+7, elements 8..15 = K kb+16..kb+23, kb = k0 + (l>=16 ? 8 : 0).
static __device__ inline v16h load_a16(const _Float16* p) {
    H16 u;
    u.h[0] = *(const v8h*)p;
    u.h[1] = *(const v8h*)(p + 16);
    return u.v;
}
// B-fragment: lane n holds column (n&15); same K interleave as A:
// elements 0..7 = K kb..kb+7, elements 8..15 = K kb+16..kb+23, kb = k0 + (n>=16 ? 8 : 0) (caller folds the +8).
static __device__ inline v16h load_b16(const _Float16* p) {
    H16 u;
    u.h[0] = *(const v8h*)p;
    u.h[1] = *(const v8h*)(p + 16);
    return u.v;
}

static __device__ inline v8f wmma16(v16h a, v16h b, v8f c) {
    return __builtin_amdgcn_wmma_f32_16x16x32_f16(false, a, false, b, (short)0, c,
                                                  false, false);
}

static __device__ inline v8f zero8() {
    v8f z = {0.f, 0.f, 0.f, 0.f, 0.f, 0.f, 0.f, 0.f};
    return z;
}

// ---------------- conversion kernels ----------------

__global__ void cvt_f32_f16(const float* __restrict__ in, _Float16* __restrict__ out,
                            long n) {
    long i = (long)blockIdx.x * blockDim.x + threadIdx.x;
    long stride = (long)gridDim.x * blockDim.x;
    for (; i < n; i += stride) out[i] = (_Float16)in[i];
}

// Wt[c*K + k] = W[k*N + c]  (transpose + downconvert; B-fragments then read contiguous K)
__global__ void cvt_w_t(const float* __restrict__ W, _Float16* __restrict__ Wt,
                        int K, int N) {
    long n = (long)K * N;
    long i = (long)blockIdx.x * blockDim.x + threadIdx.x;
    long stride = (long)gridDim.x * blockDim.x;
    for (; i < n; i += stride) {
        int k = (int)(i / N);
        int c = (int)(i % N);
        Wt[(long)c * K + k] = (_Float16)W[i];
    }
}

// ---------------- WMMA GEMM: C[M,N] = A[M,K] * Bt[N,K]^T + bias ----------------
// 128 threads = 4 waves; wave tile 32x64 (2x4 WMMA frags); workgroup tile 128x64.
// mode 0: f16 out, per-head [BH][S][64]   (Q, K)
// mode 1: f16 out, per-head transposed [BH][64][S]   (V)
// mode 2: f32 out, row-major [M][N]        (final projection)
__global__ __launch_bounds__(128) void gemm_wmma(
    const _Float16* __restrict__ A, const _Float16* __restrict__ Bt,
    const float* __restrict__ bias, void* __restrict__ out,
    int M, int N, int K, int S, int mode) {
    const int lane = threadIdx.x & 31;
    const int wave = threadIdx.x >> 5;
    const int hi   = lane >> 4;   // half-wave: 0 or 1
    const int ln   = lane & 15;

    const int row0 = blockIdx.y * 128 + wave * 32;
    const int col0 = blockIdx.x * 64;

    v8f acc[2][4];
#pragma unroll
    for (int mi = 0; mi < 2; ++mi)
#pragma unroll
        for (int nb = 0; nb < 4; ++nb) acc[mi][nb] = zero8();

    for (int k0 = 0; k0 < K; k0 += 32) {
        v16h af[2];
#pragma unroll
        for (int mi = 0; mi < 2; ++mi) {
            int row = row0 + mi * 16 + ln;
            af[mi] = load_a16(A + (size_t)row * K + k0 + hi * 8);
        }
#pragma unroll
        for (int nb = 0; nb < 4; ++nb) {
            int col = col0 + nb * 16 + ln;
            v16h bf = load_b16(Bt + (size_t)col * K + k0 + hi * 8);
#pragma unroll
            for (int mi = 0; mi < 2; ++mi)
                acc[mi][nb] = wmma16(af[mi], bf, acc[mi][nb]);
        }
    }

#pragma unroll
    for (int mi = 0; mi < 2; ++mi) {
#pragma unroll
        for (int nb = 0; nb < 4; ++nb) {
            int col = col0 + nb * 16 + ln;
            float bv = bias[col];
#pragma unroll
            for (int r = 0; r < 8; ++r) {
                int row = row0 + mi * 16 + r + hi * 8;
                float val = acc[mi][nb][r] + bv;
                if (mode == 2) {
                    ((float*)out)[(size_t)row * N + col] = val;
                } else {
                    int b = row / S, s = row - b * S;
                    int h = col >> 6, d = col & 63;
                    if (mode == 0) {
                        ((_Float16*)out)[((size_t)(b * HEADS + h) * S + s) * HEADDIM + d] =
                            (_Float16)val;
                    } else {  // mode 1: V transposed [BH][64][S]
                        ((_Float16*)out)[((size_t)(b * HEADS + h) * HEADDIM + d) * S + s] =
                            (_Float16)val;
                    }
                }
            }
        }
    }
}

// ---------------- flash attention: online softmax, WMMA QK^T and PV ----------------
// grid (S/64, B*HEADS), 128 threads = 4 waves, 16 queries per wave.
__global__ __launch_bounds__(128) void attn_flash(
    const _Float16* __restrict__ Qh, const _Float16* __restrict__ Kh,
    const _Float16* __restrict__ Vt, _Float16* __restrict__ Ah, int S) {
    const int lane = threadIdx.x & 31;
    const int wave = threadIdx.x >> 5;
    const int hi   = lane >> 4;
    const int ln   = lane & 15;

    const int bh = blockIdx.y;
    const int b  = bh >> 4;
    const int h  = bh & 15;
    const _Float16* Qp = Qh + (size_t)bh * S * HEADDIM;
    const _Float16* Kp = Kh + (size_t)bh * S * HEADDIM;
    const _Float16* Vp = Vt + (size_t)bh * HEADDIM * S;
    const int q0 = blockIdx.x * 64 + wave * 16;

    // Q A-fragments: resident in registers for the whole key loop.
    v16h qf[2];
#pragma unroll
    for (int ks = 0; ks < 2; ++ks) {
        int row = q0 + ln;
        qf[ks] = load_a16(Qp + (size_t)row * HEADDIM + ks * 32 + hi * 8);
    }

    v8f o[4];
#pragma unroll
    for (int db = 0; db < 4; ++db) o[db] = zero8();
    float mstat[8], lstat[8];
#pragma unroll
    for (int r = 0; r < 8; ++r) { mstat[r] = -1e30f; lstat[r] = 0.f; }

    __shared__ __align__(16) _Float16 Pb[4][16][72];  // per-wave P tile, padded pitch

    const float scale = 0.125f;  // 1/sqrt(64)

    for (int kc = 0; kc < S; kc += 64) {
        // ---- S tile = Q @ K^T (16 x 64) ----
        v8f sc[4];
#pragma unroll
        for (int nb = 0; nb < 4; ++nb) sc[nb] = zero8();
#pragma unroll
        for (int nb = 0; nb < 4; ++nb) {
#pragma unroll
            for (int ks = 0; ks < 2; ++ks) {
                int key = kc + nb * 16 + ln;
                v16h bf = load_b16(Kp + (size_t)key * HEADDIM + ks * 32 + hi * 8);
                sc[nb] = wmma16(qf[ks], bf, sc[nb]);
            }
        }
        // ---- online softmax (row stats live in each 16-lane half) ----
#pragma unroll
        for (int r = 0; r < 8; ++r) {
            float rm = -1e30f;
#pragma unroll
            for (int nb = 0; nb < 4; ++nb) {
                float t = sc[nb][r] * scale;
                sc[nb][r] = t;
                rm = fmaxf(rm, t);
            }
            rm = fmaxf(rm, __shfl_xor(rm, 1));
            rm = fmaxf(rm, __shfl_xor(rm, 2));
            rm = fmaxf(rm, __shfl_xor(rm, 4));
            rm = fmaxf(rm, __shfl_xor(rm, 8));
            float mn = fmaxf(mstat[r], rm);
            float corr = __expf(mstat[r] - mn);
            mstat[r] = mn;
            float rs = 0.f;
#pragma unroll
            for (int nb = 0; nb < 4; ++nb) {
                float pj = __expf(sc[nb][r] - mn);
                rs += pj;
                Pb[wave][r + hi * 8][nb * 16 + ln] = (_Float16)pj;
            }
            rs += __shfl_xor(rs, 1);
            rs += __shfl_xor(rs, 2);
            rs += __shfl_xor(rs, 4);
            rs += __shfl_xor(rs, 8);
            lstat[r] = lstat[r] * corr + rs;
#pragma unroll
            for (int db = 0; db < 4; ++db) o[db][r] = o[db][r] * corr;
        }
        // ---- O += P @ V  (P re-read from LDS in A-fragment layout) ----
#pragma unroll
        for (int db = 0; db < 4; ++db) {
#pragma unroll
            for (int ks = 0; ks < 2; ++ks) {
                const _Float16* pp = &Pb[wave][ln][ks * 32 + hi * 8];
                H16 u;
                u.h[0] = *(const v8h*)pp;
                u.h[1] = *(const v8h*)(pp + 16);
                int d = db * 16 + ln;
                v16h vf = load_b16(Vp + (size_t)d * S + kc + ks * 32 + hi * 8);
                o[db] = wmma16(u.v, vf, o[db]);
            }
        }
    }

    // ---- epilogue: normalize rows, write [B][S][H*64] f16 ----
#pragma unroll
    for (int r = 0; r < 8; ++r) {
        float inv = 1.0f / lstat[r];
        int s = q0 + r + hi * 8;
#pragma unroll
        for (int db = 0; db < 4; ++db) {
            int col = h * HEADDIM + db * 16 + ln;
            Ah[((size_t)b * S + s) * DMODEL + col] = (_Float16)(o[db][r] * inv);
        }
    }
}

extern "C" void kernel_launch(void* const* d_in, const int* in_sizes, int n_in,
                              void* d_out, int out_size, void* d_ws, size_t ws_size,
                              hipStream_t stream) {
    const float* X  = (const float*)d_in[0];
    const float* Wq = (const float*)d_in[1];
    const float* bq = (const float*)d_in[2];
    const float* Wk = (const float*)d_in[3];
    const float* bk = (const float*)d_in[4];
    const float* Wv = (const float*)d_in[5];
    const float* bv = (const float*)d_in[6];
    const float* Wo = (const float*)d_in[7];
    const float* bo = (const float*)d_in[8];
    float* out = (float*)d_out;

    const int B = 2, S = 2048, D = DMODEL;
    const int NT = B * S;  // 4096 rows

    char* ws = (char*)d_ws;
    size_t off = 0;
    _Float16* Xh  = (_Float16*)(ws + off); off += (size_t)NT * D * 2;  // 8 MiB
    _Float16* Wqh = (_Float16*)(ws + off); off += (size_t)D * D * 2;   // 2 MiB each
    _Float16* Wkh = (_Float16*)(ws + off); off += (size_t)D * D * 2;
    _Float16* Wvh = (_Float16*)(ws + off); off += (size_t)D * D * 2;
    _Float16* Woh = (_Float16*)(ws + off); off += (size_t)D * D * 2;
    _Float16* Qh  = (_Float16*)(ws + off); off += (size_t)NT * D * 2;
    _Float16* Kh  = (_Float16*)(ws + off); off += (size_t)NT * D * 2;
    _Float16* Vt  = (_Float16*)(ws + off); off += (size_t)NT * D * 2;
    _Float16* Ah  = (_Float16*)(ws + off); off += (size_t)NT * D * 2;  // total 48 MiB

    // 1) convert activations + transposed weights to f16
    long nX = (long)NT * D;
    cvt_f32_f16<<<2048, 256, 0, stream>>>(X, Xh, nX);
    cvt_w_t<<<1024, 256, 0, stream>>>(Wq, Wqh, D, D);
    cvt_w_t<<<1024, 256, 0, stream>>>(Wk, Wkh, D, D);
    cvt_w_t<<<1024, 256, 0, stream>>>(Wv, Wvh, D, D);
    cvt_w_t<<<1024, 256, 0, stream>>>(Wo, Woh, D, D);

    // 2) QKV projections (WMMA GEMMs)
    dim3 g(D / 64, NT / 128);  // (16, 32)
    gemm_wmma<<<g, 128, 0, stream>>>(Xh, Wqh, bq, Qh, NT, D, D, S, 0);
    gemm_wmma<<<g, 128, 0, stream>>>(Xh, Wkh, bk, Kh, NT, D, D, S, 0);
    gemm_wmma<<<g, 128, 0, stream>>>(Xh, Wvh, bv, Vt, NT, D, D, S, 1);

    // 3) flash attention
    attn_flash<<<dim3(S / 64, B * HEADS), 128, 0, stream>>>(Qh, Kh, Vt, Ah, S);

    // 4) output projection, f32 + bias to d_out
    gemm_wmma<<<g, 128, 0, stream>>>(Ah, Woh, bo, out, NT, D, D, S, 2);
}
